// Correspondence_Net_5970004541892
// MI455X (gfx1250) — matrix operand order repaired and checked by reading, not compiled, not run on hardware
//
#include <hip/hip_runtime.h>
#include <hip/hip_bf16.h>
#include <math.h>

// ---------------------------------------------------------------------------
// CDNA5 / gfx1250 correspondence network.
// - All GEMM-like work (3x3 convs as implicit GEMM, 1x1 convs, 4096x4096x256
//   correlation) -> v_wmma_f32_16x16x32_f16, f32->f16 conversion fused into
//   LDS staging (global b128 loads, ds b128 stores, LDS double buffering).
// - FULL template variants drop every bounds check when tiles divide exactly
//   (the dominant launches); generic variants cover conv1_1 / 64-ch tails.
// - Online single-pass softmax with non-temporal reads of the 134MB f matrix;
//   correlation writes f with non-temporal stores.
// ---------------------------------------------------------------------------

typedef _Float16 v16h __attribute__((ext_vector_type(16)));
typedef _Float16 v8h  __attribute__((ext_vector_type(8)));
typedef float    v8f  __attribute__((ext_vector_type(8)));
typedef float    v4f  __attribute__((ext_vector_type(4)));

#define LDS_LD 40   // f16 per LDS row: 32 + 8 pad (80B rows, 16B-aligned chunks)

__device__ __forceinline__ v16h frag16(const _Float16* p0, const _Float16* p1) {
    v8h a = *(const v8h*)p0;
    v8h b = *(const v8h*)p1;
    v16h r;
#pragma unroll
    for (int i = 0; i < 8; ++i) { r[i] = a[i]; r[i + 8] = b[i]; }
    return r;
}

// One wave computes a 32(M) x 64(N) patch = 2x4 WMMA 16x16 tiles.
// A frag (16x32): lane<16 -> row=lane, K {0..7,16..23}; lane>=16 -> K {8..15,24..31}.
// B frag (32x16): lane<16 -> col=lane, K {0..7,16..23}; lane>=16 -> col=lane-16, K {8..15,24..31}.
// lsB stored [n][k] so both fragments are contiguous 16B chunks.
__device__ __forceinline__ void wmma_block(const _Float16* lsA, const _Float16* lsB,
                                           v8f (&acc)[2][4], int wm, int wn,
                                           int ll, int hi) {
#pragma unroll
    for (int mi = 0; mi < 2; ++mi) {
        const _Float16* pa = lsA + (wm + mi * 16 + ll) * LDS_LD + hi * 8;
        v16h af = frag16(pa, pa + 16);
#pragma unroll
        for (int ni = 0; ni < 4; ++ni) {
            const _Float16* pb = lsB + (wn + ni * 16 + ll) * LDS_LD + hi * 8;
            v16h bf = frag16(pb, pb + 16);
            acc[mi][ni] = __builtin_amdgcn_wmma_f32_16x16x32_f16(
                false, af, false, bf, (short)0, acc[mi][ni], false, false);
        }
    }
}

// ---------------------------------------------------------------------------
// Implicit-GEMM 3x3 conv. PAD: 0 = zero-pad(1), 1 = reflect-pad(1).
// FULL: Cout%128==0, K%32==0, npix%128==0 -> no bounds checks anywhere.
// ---------------------------------------------------------------------------
template <int PAD, bool FULL>
__global__ __launch_bounds__(256) void conv3x3_wmma_k(
    const float* __restrict__ x, const float* __restrict__ w,
    const float* __restrict__ bias, float* __restrict__ out,
    int Cin, int H, int W, int Cout, int Ho, int Wo,
    int stride, int relu, size_t xB, size_t oB) {
    const int npix = Ho * Wo;
    const int K = Cin * 9;
    const int z = blockIdx.z;
    const int tileN = blockIdx.x * 128;
    const int tileM = blockIdx.y * 128;
    x += (size_t)z * xB;
    out += (size_t)z * oB;

    __shared__ __align__(16) _Float16 lsA[2][128 * LDS_LD];
    __shared__ __align__(16) _Float16 lsB[2][128 * LDS_LD];

    const int tid = threadIdx.x;
    const int lane = tid & 31;
    const int wv = tid >> 5;
    const int wm = (wv >> 1) * 32;
    const int wn = (wv & 1) * 64;
    const int ll = lane & 15;
    const int hi = lane >> 4;

    const int rowA = tid >> 1;           // staged Cout row
    const int kofA = (tid & 1) * 16;
    const int pixN = tileN + (tid >> 1); // staged pixel
    const int kofB = (tid & 1) * 16;
    const bool k4 = (K & 3) == 0;

    v4f aV[4];
    float bR[16];

    auto loadA = [&](int k0) {
        int gm = tileM + rowA;
#pragma unroll
        for (int c = 0; c < 4; ++c) {
            int k = k0 + kofA + c * 4;
            if (FULL) {
                aV[c] = *(const v4f*)&w[(size_t)gm * K + k];
            } else {
                v4f v = {};
                if (gm < Cout) {
                    if (k4 && k + 4 <= K) {
                        v = *(const v4f*)&w[(size_t)gm * K + k];
                    } else {
#pragma unroll
                        for (int j = 0; j < 4; ++j)
                            if (k + j < K) v[j] = w[(size_t)gm * K + k + j];
                    }
                }
                aV[c] = v;
            }
        }
    };
    auto storeA = [&](int buf) {
#pragma unroll
        for (int c2 = 0; c2 < 2; ++c2) {
            v8h h;
#pragma unroll
            for (int j = 0; j < 8; ++j) h[j] = (_Float16)aV[c2 * 2 + (j >> 2)][j & 3];
            *(v8h*)&lsA[buf][rowA * LDS_LD + kofA + c2 * 8] = h;
        }
    };

    const int oh = pixN / Wo, ow = pixN - oh * Wo;
    const bool okpix = FULL || (pixN < npix);
    auto loadB = [&](int k0) {
#pragma unroll
        for (int i = 0; i < 16; ++i) {
            int k = k0 + kofB + i;
            float v = 0.f;
            if (okpix && (FULL || k < K)) {
                int ci = k / 9, r = k - ci * 9;
                int kh = r / 3, kw = r - kh * 3;
                int ih = oh * stride + kh - 1;
                int iw = ow * stride + kw - 1;
                if (PAD == 1) {
                    ih = (ih < 0) ? -ih : (ih >= H ? 2 * H - 2 - ih : ih);
                    iw = (iw < 0) ? -iw : (iw >= W ? 2 * W - 2 - iw : iw);
                    v = x[((size_t)ci * H + ih) * W + iw];
                } else {
                    if (ih >= 0 && ih < H && iw >= 0 && iw < W)
                        v = x[((size_t)ci * H + ih) * W + iw];
                }
            }
            bR[i] = v;
        }
    };
    auto storeB = [&](int buf) {
#pragma unroll
        for (int c2 = 0; c2 < 2; ++c2) {
            v8h h;
#pragma unroll
            for (int j = 0; j < 8; ++j) h[j] = (_Float16)bR[c2 * 8 + j];
            *(v8h*)&lsB[buf][(tid >> 1) * LDS_LD + kofB + c2 * 8] = h;
        }
    };

    v8f acc[2][4];
#pragma unroll
    for (int i = 0; i < 2; ++i)
#pragma unroll
        for (int j = 0; j < 4; ++j) { v8f zz = {}; acc[i][j] = zz; }

    loadA(0);
    loadB(0);
    storeA(0);
    storeB(0);
    __syncthreads();
    int cur = 0;
    for (int k0 = 32; k0 < K; k0 += 32) {
        loadA(k0);                 // next-tile global loads overlap WMMA
        loadB(k0);
        wmma_block(lsA[cur], lsB[cur], acc, wm, wn, ll, hi);
        storeA(cur ^ 1);
        storeB(cur ^ 1);
        __syncthreads();
        cur ^= 1;
    }
    wmma_block(lsA[cur], lsB[cur], acc, wm, wn, ll, hi);

#pragma unroll
    for (int mi = 0; mi < 2; ++mi)
#pragma unroll
        for (int ni = 0; ni < 4; ++ni) {
            int n = tileN + wn + ni * 16 + ll;
            if (!FULL && n >= npix) continue;
#pragma unroll
            for (int r = 0; r < 8; ++r) {
                int m = tileM + wm + mi * 16 + r + hi * 8;
                if (FULL || m < Cout) {
                    float v = acc[mi][ni][r] + bias[m];
                    if (relu) v = v > 0.f ? v : 0.f;
                    out[(size_t)m * npix + n] = v;
                }
            }
        }
}

// ---------------------------------------------------------------------------
// General WMMA GEMM: C[M,N] = A x B (+bias). TA: A[k*lda+m]; TB: B[n*ldb+k].
// FULL: M,N%128==0, K%32==0 -> branch-free loaders and epilogue.
// ntout -> non-temporal C stores.
// ---------------------------------------------------------------------------
template <bool TA, bool TB, bool FULL>
__global__ __launch_bounds__(256) void gemm_wmma_k(
    const float* __restrict__ A, const float* __restrict__ B,
    const float* __restrict__ bias, float* __restrict__ C,
    int M, int N, int Kd, int lda, int ldb, int ldc, int ntout,
    size_t sA, size_t sB, size_t sC) {
    const int z = blockIdx.z;
    A += (size_t)z * sA;
    B += (size_t)z * sB;
    C += (size_t)z * sC;
    const int tileN = blockIdx.x * 128;
    const int tileM = blockIdx.y * 128;

    __shared__ __align__(16) _Float16 lsA[2][128 * LDS_LD];
    __shared__ __align__(16) _Float16 lsB[2][128 * LDS_LD];

    const int tid = threadIdx.x;
    const int lane = tid & 31;
    const int wv = tid >> 5;
    const int wm = (wv >> 1) * 32;
    const int wn = (wv & 1) * 64;
    const int ll = lane & 15;
    const int hi = lane >> 4;

    v4f aV[4], bV[4];

    auto loadA = [&](int k0) {
        if constexpr (!TA) {
            int gm = tileM + (tid >> 1);
            int kb = k0 + (tid & 1) * 16;
#pragma unroll
            for (int c = 0; c < 4; ++c) {
                int k = kb + c * 4;
                if (FULL) {
                    aV[c] = *(const v4f*)&A[(size_t)gm * lda + k];
                } else {
                    v4f v = {};
                    if (gm < M) {
                        if (k + 4 <= Kd) v = *(const v4f*)&A[(size_t)gm * lda + k];
                        else {
#pragma unroll
                            for (int j = 0; j < 4; ++j)
                                if (k + j < Kd) v[j] = A[(size_t)gm * lda + k + j];
                        }
                    }
                    aV[c] = v;
                }
            }
        } else {
            // transposed A: contiguous along m -> vector loads of 4 rows
#pragma unroll
            for (int c = 0; c < 4; ++c) {
                int t = tid + c * 256;
                int krel = t & 31;
                int m0 = (t >> 5) * 4;
                int k = k0 + krel;
                int gm0 = tileM + m0;
                if (FULL) {
                    aV[c] = *(const v4f*)&A[(size_t)k * lda + gm0];
                } else {
                    v4f v = {};
                    if (k < Kd) {
                        if (gm0 + 4 <= M) v = *(const v4f*)&A[(size_t)k * lda + gm0];
                        else {
#pragma unroll
                            for (int j = 0; j < 4; ++j)
                                if (gm0 + j < M) v[j] = A[(size_t)k * lda + gm0 + j];
                        }
                    }
                    aV[c] = v;
                }
            }
        }
    };
    auto storeA = [&](int buf) {
        if constexpr (!TA) {
            int m = tid >> 1, kb = (tid & 1) * 16;
#pragma unroll
            for (int c2 = 0; c2 < 2; ++c2) {
                v8h h;
#pragma unroll
                for (int j = 0; j < 8; ++j) h[j] = (_Float16)aV[c2 * 2 + (j >> 2)][j & 3];
                *(v8h*)&lsA[buf][m * LDS_LD + kb + c2 * 8] = h;
            }
        } else {
#pragma unroll
            for (int c = 0; c < 4; ++c) {
                int t = tid + c * 256;
                int krel = t & 31;
                int m0 = (t >> 5) * 4;
#pragma unroll
                for (int j = 0; j < 4; ++j)
                    lsA[buf][(m0 + j) * LDS_LD + krel] = (_Float16)aV[c][j];
            }
        }
    };
    auto loadB = [&](int k0) {
        if constexpr (!TB) {
            // B contiguous along n -> vector loads of 4 columns
#pragma unroll
            for (int c = 0; c < 4; ++c) {
                int t = tid + c * 256;
                int krel = t & 31;
                int n0 = (t >> 5) * 4;
                int k = k0 + krel;
                int gn0 = tileN + n0;
                if (FULL) {
                    bV[c] = *(const v4f*)&B[(size_t)k * ldb + gn0];
                } else {
                    v4f v = {};
                    if (k < Kd) {
                        if (gn0 + 4 <= N) v = *(const v4f*)&B[(size_t)k * ldb + gn0];
                        else {
#pragma unroll
                            for (int j = 0; j < 4; ++j)
                                if (gn0 + j < N) v[j] = B[(size_t)k * ldb + gn0 + j];
                        }
                    }
                    bV[c] = v;
                }
            }
        } else {
            int gn = tileN + (tid >> 1);
            int kb = k0 + (tid & 1) * 16;
#pragma unroll
            for (int c = 0; c < 4; ++c) {
                int k = kb + c * 4;
                if (FULL) {
                    bV[c] = *(const v4f*)&B[(size_t)gn * ldb + k];
                } else {
                    v4f v = {};
                    if (gn < N) {
                        if (k + 4 <= Kd) v = *(const v4f*)&B[(size_t)gn * ldb + k];
                        else {
#pragma unroll
                            for (int j = 0; j < 4; ++j)
                                if (k + j < Kd) v[j] = B[(size_t)gn * ldb + k + j];
                        }
                    }
                    bV[c] = v;
                }
            }
        }
    };
    auto storeB = [&](int buf) {
        if constexpr (!TB) {
#pragma unroll
            for (int c = 0; c < 4; ++c) {
                int t = tid + c * 256;
                int krel = t & 31;
                int n0 = (t >> 5) * 4;
#pragma unroll
                for (int j = 0; j < 4; ++j)
                    lsB[buf][(n0 + j) * LDS_LD + krel] = (_Float16)bV[c][j];
            }
        } else {
            int n = tid >> 1, kb = (tid & 1) * 16;
#pragma unroll
            for (int c2 = 0; c2 < 2; ++c2) {
                v8h h;
#pragma unroll
                for (int j = 0; j < 8; ++j) h[j] = (_Float16)bV[c2 * 2 + (j >> 2)][j & 3];
                *(v8h*)&lsB[buf][n * LDS_LD + kb + c2 * 8] = h;
            }
        }
    };

    v8f acc[2][4];
#pragma unroll
    for (int i = 0; i < 2; ++i)
#pragma unroll
        for (int j = 0; j < 4; ++j) { v8f zz = {}; acc[i][j] = zz; }

    loadA(0);
    loadB(0);
    storeA(0);
    storeB(0);
    __syncthreads();
    int cur = 0;
    for (int k0 = 32; k0 < Kd; k0 += 32) {
        loadA(k0);
        loadB(k0);
        wmma_block(lsA[cur], lsB[cur], acc, wm, wn, ll, hi);
        storeA(cur ^ 1);
        storeB(cur ^ 1);
        __syncthreads();
        cur ^= 1;
    }
    wmma_block(lsA[cur], lsB[cur], acc, wm, wn, ll, hi);

#pragma unroll
    for (int mi = 0; mi < 2; ++mi)
#pragma unroll
        for (int ni = 0; ni < 4; ++ni) {
            int n = tileN + wn + ni * 16 + ll;
            if (!FULL && n >= N) continue;
#pragma unroll
            for (int r = 0; r < 8; ++r) {
                int m = tileM + wm + mi * 16 + r + hi * 8;
                if (FULL || m < M) {
                    float v = acc[mi][ni][r];
                    if (bias) v += bias[m];
                    float* dst = &C[(size_t)m * ldc + n];
                    if (ntout) __builtin_nontemporal_store(v, dst);
                    else *dst = v;
                }
            }
        }
}

// ---------------------------------------------------------------------------
// Elementwise / reduction kernels
// ---------------------------------------------------------------------------
__global__ void preprocess_k(const float* __restrict__ in, float* __restrict__ out,
                             int HW, int total) {
    int idx = blockIdx.x * 256 + threadIdx.x;
    if (idx >= total) return;
    const float mean[3] = {0.40760392f, 0.45795686f, 0.48501961f};
    int p = idx % HW;
    int c = (idx / HW) % 3;
    int b = idx / (3 * HW);
    out[idx] = (in[((size_t)b * 3 + (2 - c)) * HW + p] - mean[c]) * 255.f;
}

__global__ void maxpool2_k(const float* __restrict__ in, float* __restrict__ out,
                           int Ctot, int H, int W) {
    int Ho = H / 2, Wo = W / 2;
    int total = Ctot * Ho * Wo;
    int idx = blockIdx.x * 256 + threadIdx.x;
    if (idx >= total) return;
    int w = idx % Wo, h = (idx / Wo) % Ho, c = idx / (Wo * Ho);
    const float* p = in + ((size_t)c * H + 2 * h) * W + 2 * w;
    out[idx] = fmaxf(fmaxf(p[0], p[1]), fmaxf(p[W], p[W + 1]));
}

__global__ void avgpool4_k(const float* __restrict__ in, float* __restrict__ out,
                           int Ctot, int H, int W) {
    int Ho = H / 4, Wo = W / 4;
    int total = Ctot * Ho * Wo;
    int idx = blockIdx.x * 256 + threadIdx.x;
    if (idx >= total) return;
    int w = idx % Wo, h = (idx / Wo) % Ho, c = idx / (Wo * Ho);
    const float* p = in + ((size_t)c * H + 4 * h) * W + 4 * w;
    float s = 0.f;
#pragma unroll
    for (int i = 0; i < 4; ++i)
#pragma unroll
        for (int j = 0; j < 4; ++j) s += p[(size_t)i * W + j];
    out[idx] = s * (1.f / 16.f);
}

__global__ void upsample_k(const float* __restrict__ in, float* __restrict__ out,
                           int Bn, int C, int Hin, int Win, int s,
                           size_t inB, size_t outB) {
    int Ho = Hin * s, Wo = Win * s;
    int total = Bn * C * Ho * Wo;
    int idx = blockIdx.x * 256 + threadIdx.x;
    if (idx >= total) return;
    int w = idx % Wo, h = (idx / Wo) % Ho;
    int c = (idx / (Wo * Ho)) % C, b = idx / (Wo * Ho * C);
    out[(size_t)b * outB + ((size_t)c * Ho + h) * Wo + w] =
        in[(size_t)b * inB + ((size_t)c * Hin + (h / s)) * Win + (w / s)];
}

// Instance norm + optional residual + optional PReLU, in-place. Block per (b,c).
__global__ __launch_bounds__(256) void inorm_prelu_k(
    float* __restrict__ xio, const float* __restrict__ res,
    const float* __restrict__ alpha, int C, int npix,
    size_t bstride, size_t res_bstride) {
    int bc = blockIdx.x;
    int b = bc / C, c = bc % C;
    float* p = xio + (size_t)b * bstride + (size_t)c * npix;
    const float* r = res ? res + (size_t)b * res_bstride + (size_t)c * npix : nullptr;
    __shared__ float s1[256], s2[256];
    int tid = threadIdx.x;
    float sum = 0.f, sq = 0.f;
    for (int i = tid; i < npix; i += 256) {
        float v = p[i];
        sum += v;
        sq += v * v;
    }
    s1[tid] = sum;
    s2[tid] = sq;
    __syncthreads();
    for (int o = 128; o > 0; o >>= 1) {
        if (tid < o) { s1[tid] += s1[tid + o]; s2[tid] += s2[tid + o]; }
        __syncthreads();
    }
    float m = s1[0] / (float)npix;
    float var = s2[0] / (float)npix - m * m;
    float rinv = rsqrtf(var + 1e-5f);
    float a = alpha ? alpha[0] : 0.f;
    for (int i = tid; i < npix; i += 256) {
        float v = (p[i] - m) * rinv;
        if (r) v += r[i];
        if (alpha) v = v >= 0.f ? v : a * v;
        p[i] = v;
    }
}

__global__ __launch_bounds__(256) void rowmean_sub_k(float* __restrict__ x,
                                                     int C, int N, size_t bstride) {
    int bc = blockIdx.x;
    int b = bc / C, c = bc % C;
    float* p = x + (size_t)b * bstride + (size_t)c * N;
    __shared__ float s1[256];
    int tid = threadIdx.x;
    float sum = 0.f;
    for (int i = tid; i < N; i += 256) sum += p[i];
    s1[tid] = sum;
    __syncthreads();
    for (int o = 128; o > 0; o >>= 1) {
        if (tid < o) s1[tid] += s1[tid + o];
        __syncthreads();
    }
    float m = s1[0] / (float)N;
    for (int i = tid; i < N; i += 256) p[i] -= m;
}

__global__ __launch_bounds__(256) void l2norm_ch_k(float* __restrict__ x,
                                                   int Bn, int C, int N,
                                                   size_t bstride) {
    int idx = blockIdx.x * 256 + threadIdx.x;
    if (idx >= Bn * N) return;
    int b = idx / N, n = idx % N;
    float* base = x + (size_t)b * bstride + n;
    float s = 0.f;
    for (int c = 0; c < C; ++c) {
        float v = base[(size_t)c * N];
        s += v * v;
    }
    float inv = 1.f / (sqrtf(s) + 2.220446049250313e-16f);
    for (int c = 0; c < C; ++c) base[(size_t)c * N] *= inv;
}

// Online softmax + attention readout + sim, single pass over each f row.
__global__ __launch_bounds__(256) void softmax_attend_k(
    const float* __restrict__ f, const float* __restrict__ Bl,
    float* __restrict__ y, float* __restrict__ sim, int N, int M) {
    int n = blockIdx.x;
    int b = blockIdx.y;
    const float* row = f + ((size_t)b * N + n) * M;
    const float* bl = Bl + (size_t)b * 3 * M;
    int tid = threadIdx.x;
    const float invT = 200.f;  // 1 / 0.005

    float mk = -1e30f, se = 0.f, w0 = 0.f, w1 = 0.f, w2 = 0.f;
    for (int m = tid; m < M; m += 256) {
        float v = __builtin_nontemporal_load(&row[m]);
        float b0 = bl[m], b1 = bl[M + m], b2 = bl[2 * M + m];
        if (v > mk) {
            float s = __expf((mk - v) * invT);
            se = se * s + 1.f;
            w0 = w0 * s + b0;
            w1 = w1 * s + b1;
            w2 = w2 * s + b2;
            mk = v;
        } else {
            float e = __expf((v - mk) * invT);
            se += e;
            w0 += e * b0;
            w1 += e * b1;
            w2 += e * b2;
        }
    }

    __shared__ float sm[256], ss[256], s0[256], s1[256], s2[256];
    sm[tid] = mk; ss[tid] = se; s0[tid] = w0; s1[tid] = w1; s2[tid] = w2;
    __syncthreads();
    for (int o = 128; o > 0; o >>= 1) {
        if (tid < o) {
            float ma = sm[tid], mb = sm[tid + o];
            float Mx = fmaxf(ma, mb);
            float ea = __expf((ma - Mx) * invT);
            float eb = __expf((mb - Mx) * invT);
            sm[tid] = Mx;
            ss[tid] = ss[tid] * ea + ss[tid + o] * eb;
            s0[tid] = s0[tid] * ea + s0[tid + o] * eb;
            s1[tid] = s1[tid] * ea + s1[tid + o] * eb;
            s2[tid] = s2[tid] * ea + s2[tid + o] * eb;
        }
        __syncthreads();
    }
    if (tid == 0) {
        float inv = 1.f / ss[0];
        sim[(size_t)b * N + n] = sm[0];
        y[((size_t)b * 3 + 0) * N + n] = s0[0] * inv;
        y[((size_t)b * 3 + 1) * N + n] = s1[0] * inv;
        y[((size_t)b * 3 + 2) * N + n] = s2[0] * inv;
    }
}

// ---------------------------------------------------------------------------
// Host orchestration
// ---------------------------------------------------------------------------
static inline int cdiv(int a, int b) { return (a + b - 1) / b; }

extern "C" void kernel_launch(void* const* d_in, const int* in_sizes, int n_in,
                              void* d_out, int out_size, void* d_ws, size_t ws_size,
                              hipStream_t stream) {
    const int Bn = 2;
    auto F = [&](int i) { return (const float*)d_in[i]; };

    float* ws = (float*)d_ws;
    size_t off = 0;
    auto alloc = [&](size_t n) {
        float* p = ws + off;
        off += (n + 63) & ~(size_t)63;
        return p;
    };
    float* ping  = alloc((size_t)Bn * 64 * 256 * 256);
    float* pong  = alloc((size_t)Bn * 64 * 256 * 256);
    float* r21   = alloc((size_t)Bn * 128 * 128 * 128);
    float* r31   = alloc((size_t)Bn * 256 * 64 * 64);
    float* r41   = alloc((size_t)Bn * 512 * 32 * 32);
    float* r51   = alloc((size_t)Bn * 512 * 16 * 16);
    float* hcat  = alloc((size_t)Bn * 256 * 64 * 64);
    float* t1b   = alloc((size_t)Bn * 256 * 64 * 64);
    float* t2b   = alloc((size_t)Bn * 256 * 64 * 64);
    float* atmp1 = alloc((size_t)Bn * 128 * 128 * 128);
    float* atmp2 = alloc((size_t)Bn * 256 * 32 * 32);
    float* theta = alloc((size_t)Bn * 256 * 4096);
    float* phi   = alloc((size_t)Bn * 256 * 4096);
    float* fbuf  = alloc((size_t)Bn * 4096 * 4096);
    float* Blb   = alloc((size_t)Bn * 3 * 4096);
    float* ybuf  = alloc((size_t)Bn * 3 * 4096);
    float* simb  = alloc((size_t)Bn * 4096);

    auto conv = [&](int pad, const float* x, const float* w, const float* b,
                    float* y, int Cin, int H, int W, int Cout, int stride,
                    int relu, size_t xB, size_t oB) {
        int Ho = (H - 1) / stride + 1;
        int Wo = (W - 1) / stride + 1;
        int npix = Ho * Wo;
        int K = Cin * 9;
        bool full = (Cout % 128 == 0) && (K % 32 == 0) && (npix % 128 == 0);
        dim3 g(cdiv(npix, 128), cdiv(Cout, 128), Bn);
        if (pad) {
            if (full)
                conv3x3_wmma_k<1, true><<<g, 256, 0, stream>>>(
                    x, w, b, y, Cin, H, W, Cout, Ho, Wo, stride, relu, xB, oB);
            else
                conv3x3_wmma_k<1, false><<<g, 256, 0, stream>>>(
                    x, w, b, y, Cin, H, W, Cout, Ho, Wo, stride, relu, xB, oB);
        } else {
            if (full)
                conv3x3_wmma_k<0, true><<<g, 256, 0, stream>>>(
                    x, w, b, y, Cin, H, W, Cout, Ho, Wo, stride, relu, xB, oB);
            else
                conv3x3_wmma_k<0, false><<<g, 256, 0, stream>>>(
                    x, w, b, y, Cin, H, W, Cout, Ho, Wo, stride, relu, xB, oB);
        }
    };
    auto inorm = [&](float* x, const float* res, const float* a, int C, int npix,
                     size_t bstride, size_t res_bstride) {
        inorm_prelu_k<<<Bn * C, 256, 0, stream>>>(x, res, a, C, npix, bstride,
                                                  res_bstride);
    };
    auto mpool = [&](const float* x, float* y, int C, int H) {
        int total = Bn * C * (H / 2) * (H / 2);
        maxpool2_k<<<cdiv(total, 256), 256, 0, stream>>>(x, y, Bn * C, H, H);
    };
    auto ups = [&](const float* x, float* y, int C, int Hin, int s,
                   size_t inB, size_t outB) {
        int total = Bn * C * Hin * s * Hin * s;
        upsample_k<<<cdiv(total, 256), 256, 0, stream>>>(x, y, Bn, C, Hin, Hin, s,
                                                         inB, outB);
    };

    auto run_stream = [&](const float* rgb, float* proj, int w1x1, int b1x1) {
        int HW = 256 * 256;
        preprocess_k<<<cdiv(Bn * 3 * HW, 256), 256, 0, stream>>>(rgb, ping, HW,
                                                                 Bn * 3 * HW);
        size_t s64 = (size_t)64 * 65536, s128 = (size_t)128 * 16384;
        size_t s256 = (size_t)256 * 4096, s512 = (size_t)512 * 1024;
        conv(0, ping, F(3), F(4), pong, 3, 256, 256, 64, 1, 1, (size_t)3 * HW, s64);
        conv(0, pong, F(5), F(6), ping, 64, 256, 256, 64, 1, 1, s64, s64);
        mpool(ping, pong, 64, 256);
        conv(0, pong, F(7), F(8), r21, 64, 128, 128, 128, 1, 1, (size_t)64 * 16384, s128);
        conv(0, r21, F(9), F(10), ping, 128, 128, 128, 128, 1, 1, s128, s128);
        mpool(ping, pong, 128, 128);
        conv(0, pong, F(11), F(12), r31, 128, 64, 64, 256, 1, 1, (size_t)128 * 4096, s256);
        conv(0, r31, F(13), F(14), ping, 256, 64, 64, 256, 1, 1, s256, s256);
        conv(0, ping, F(15), F(16), pong, 256, 64, 64, 256, 1, 1, s256, s256);
        conv(0, pong, F(17), F(18), ping, 256, 64, 64, 256, 1, 1, s256, s256);
        mpool(ping, pong, 256, 64);
        conv(0, pong, F(19), F(20), r41, 256, 32, 32, 512, 1, 1, (size_t)256 * 1024, s512);
        conv(0, r41, F(21), F(22), ping, 512, 32, 32, 512, 1, 1, s512, s512);
        conv(0, ping, F(23), F(24), pong, 512, 32, 32, 512, 1, 1, s512, s512);
        conv(0, pong, F(25), F(26), ping, 512, 32, 32, 512, 1, 1, s512, s512);
        mpool(ping, pong, 512, 32);
        conv(0, pong, F(27), F(28), r51, 512, 16, 16, 512, 1, 1, (size_t)512 * 256,
             (size_t)512 * 256);

        size_t catB = (size_t)256 * 4096;
        // l2 (29..34)
        conv(1, r21, F(29), F(30), atmp1, 128, 128, 128, 128, 1, 0, s128, s128);
        inorm(atmp1, nullptr, F(31), 128, 16384, s128, 0);
        conv(1, atmp1, F(32), F(33), hcat, 128, 128, 128, 64, 2, 0, s128, catB);
        inorm(hcat, nullptr, F(34), 64, 4096, catB, 0);
        // l3 (35..40)
        conv(1, r31, F(35), F(36), atmp1, 256, 64, 64, 128, 1, 0, s256,
             (size_t)128 * 4096);
        inorm(atmp1, nullptr, F(37), 128, 4096, (size_t)128 * 4096, 0);
        conv(1, atmp1, F(38), F(39), hcat + (size_t)64 * 4096, 128, 64, 64, 64, 1, 0,
             (size_t)128 * 4096, catB);
        inorm(hcat + (size_t)64 * 4096, nullptr, F(40), 64, 4096, catB, 0);
        // l4 (41..46)
        conv(1, r41, F(41), F(42), atmp1, 512, 32, 32, 256, 1, 0, s512,
             (size_t)256 * 1024);
        inorm(atmp1, nullptr, F(43), 256, 1024, (size_t)256 * 1024, 0);
        conv(1, atmp1, F(44), F(45), atmp2, 256, 32, 32, 64, 1, 0,
             (size_t)256 * 1024, (size_t)64 * 1024);
        inorm(atmp2, nullptr, F(46), 64, 1024, (size_t)64 * 1024, 0);
        ups(atmp2, hcat + (size_t)128 * 4096, 64, 32, 2, (size_t)64 * 1024, catB);
        // l5 (47..52)
        conv(1, r51, F(47), F(48), atmp1, 512, 16, 16, 256, 1, 0,
             (size_t)512 * 256, (size_t)256 * 256);
        inorm(atmp1, nullptr, F(49), 256, 256, (size_t)256 * 256, 0);
        ups(atmp1, atmp2, 256, 16, 2, (size_t)256 * 256, (size_t)256 * 1024);
        conv(1, atmp2, F(50), F(51), atmp1, 256, 32, 32, 64, 1, 0,
             (size_t)256 * 1024, (size_t)64 * 1024);
        inorm(atmp1, nullptr, F(52), 64, 1024, (size_t)64 * 1024, 0);
        ups(atmp1, hcat + (size_t)192 * 4096, 64, 32, 2, (size_t)64 * 1024, catB);

        float* h = hcat;
        float* ta = t1b;
        float* tb = t2b;
        for (int rbi = 0; rbi < 3; ++rbi) {
            int base = 53 + 5 * rbi;
            conv(1, h, F(base + 0), F(base + 1), ta, 256, 64, 64, 256, 1, 0, catB, catB);
            inorm(ta, nullptr, F(base + 4), 256, 4096, catB, 0);
            conv(1, ta, F(base + 2), F(base + 3), tb, 256, 64, 64, 256, 1, 0, catB, catB);
            inorm(tb, h, F(base + 4), 256, 4096, catB, catB);
            float* tmp = h; h = tb; tb = tmp;
        }

        {
            dim3 g(cdiv(4096, 128), cdiv(256, 128), Bn);
            gemm_wmma_k<false, false, true><<<g, 256, 0, stream>>>(
                F(w1x1), h, F(b1x1), proj, 256, 4096, 256, 256, 4096, 4096, 0,
                0, catB, (size_t)256 * 4096);
        }
        rowmean_sub_k<<<Bn * 256, 256, 0, stream>>>(proj, 256, 4096,
                                                    (size_t)256 * 4096);
        l2norm_ch_k<<<cdiv(Bn * 4096, 256), 256, 0, stream>>>(
            proj, Bn, 256, 4096, (size_t)256 * 4096);
    };

    run_stream(F(0), theta, 68, 69);
    run_stream(F(1), phi, 70, 71);

    // correlation f[b][n][m] = sum_c theta[c][n] * phi[c][m]; NT stores for f.
    {
        dim3 g(cdiv(4096, 128), cdiv(4096, 128), Bn);
        gemm_wmma_k<true, false, true><<<g, 256, 0, stream>>>(
            theta, phi, nullptr, fbuf, 4096, 4096, 256, 4096, 4096, 4096, 1,
            (size_t)256 * 4096, (size_t)256 * 4096, (size_t)4096 * 4096);
    }

    {
        int total = Bn * 3 * 64 * 64;
        avgpool4_k<<<cdiv(total, 256), 256, 0, stream>>>(F(2), Blb, Bn * 3, 256, 256);
    }
    {
        dim3 g(4096, Bn);
        softmax_attend_k<<<g, 256, 0, stream>>>(fbuf, Blb, ybuf, simb, 4096, 4096);
    }

    float* outp = (float*)d_out;
    upsample_k<<<cdiv(Bn * 3 * 256 * 256, 256), 256, 0, stream>>>(
        ybuf, outp, Bn, 3, 64, 64, 4, (size_t)3 * 4096, (size_t)3 * 65536);
    upsample_k<<<cdiv(Bn * 1 * 256 * 256, 256), 256, 0, stream>>>(
        simb, outp + (size_t)Bn * 3 * 65536, Bn, 1, 64, 64, 4, (size_t)4096,
        (size_t)65536);
}
